// LSTM_CRF_10660108829393
// MI455X (gfx1250) — matrix operand order repaired and checked by reading, not compiled, not run on hardware
//
#include <hip/hip_runtime.h>
#include <hip/hip_bf16.h>

typedef __attribute__((ext_vector_type(16))) _Float16 v16h;
typedef __attribute__((ext_vector_type(8)))  _Float16 v8h;
typedef __attribute__((ext_vector_type(8)))  float    v8f;

#define BB   128   // batch
#define TT   512   // time
#define EE   128   // embed
#define HH   128   // hidden per dir
#define G4   512   // 4*H
#define NTAG 9
#define VV   32000

static __device__ __forceinline__ v8f wmma_f16(v16h a, v16h b, v8f c) {
    return __builtin_amdgcn_wmma_f32_16x16x32_f16(false, a, false, b, (short)0, c, false, false);
}

static __device__ __forceinline__ v16h pack16(v8h lo, v8h hi) {
    v16h r;
    #pragma unroll
    for (int j = 0; j < 8; ++j) { r[j] = lo[j]; r[8 + j] = hi[j]; }
    return r;
}

// ---------------------------------------------------------------------------
// One-time f32 -> f16 conversion sweep (embedding + weights).
// ---------------------------------------------------------------------------
__global__ __launch_bounds__(256)
void cvt_f16_kernel(const float* __restrict__ src, _Float16* __restrict__ dst, int n)
{
    int i = blockIdx.x * 256 + threadIdx.x;
    if (i < n) dst[i] = (_Float16)src[i];
}

// Pad w_out (9x256) to 16x256 f16 (rows >=9 are zero).
__global__ __launch_bounds__(256)
void wout_pad_kernel(const float* __restrict__ w_out, _Float16* __restrict__ woutP)
{
    int i = blockIdx.x * 256 + threadIdx.x;        // 0..4095
    int n = i >> 8, k = i & 255;
    woutP[i] = (n < NTAG) ? (_Float16)w_out[n * 256 + k] : (_Float16)0.f;
}

// ---------------------------------------------------------------------------
// Kernel 1: embedding gather + input projection, both directions.
// Output xp is stored TILE-SWIZZLED: [mtile][ntile][lane][8] f16, i.e. the
// exact per-lane C-fragment layout the recurrent kernel consumes (b128 I/O).
// grid=(4096,2), block=256 (8 waves); wave handles 4 n-tiles.
// ---------------------------------------------------------------------------
__global__ __launch_bounds__(256)
void proj_kernel(const int* __restrict__ sentence,
                 const _Float16* __restrict__ embH,
                 const _Float16* __restrict__ wihFH, const float* __restrict__ b_f,
                 const _Float16* __restrict__ wihBH, const float* __restrict__ b_b,
                 _Float16* __restrict__ xpF, _Float16* __restrict__ xpB)
{
    const int mtile = blockIdx.x;
    const int dir   = blockIdx.y;
    const int wv    = threadIdx.x >> 5;
    const int lane  = threadIdx.x & 31;
    const int lm    = lane & 15;
    const int lg    = lane >> 4;

    const _Float16* wih  = dir ? wihBH : wihFH;
    const float*    bias = dir ? b_b   : b_f;
    _Float16*       xp   = dir ? xpB   : xpF;

    const int r   = mtile * 16 + lm;            // r = t*B + b
    const int t   = r >> 7;
    const int b   = r & 127;
    const int tok = sentence[b * TT + t];
    const _Float16* erow = embH + (size_t)tok * EE;

    v16h A[4];
    #pragma unroll
    for (int kc = 0; kc < 4; ++kc) {
        v8h lo = *(const v8h*)(erow + kc * 32 + lg * 8);
        v8h hi = *(const v8h*)(erow + kc * 32 + 16 + lg * 8);
        A[kc] = pack16(lo, hi);
    }

    #pragma unroll
    for (int q = 0; q < 4; ++q) {
        const int nt = wv * 4 + q;
        const int n  = nt * 16 + lm;
        v8f acc = {};
        #pragma unroll
        for (int kc = 0; kc < 4; ++kc) {
            const _Float16* wr = wih + (size_t)n * EE + kc * 32 + lg * 8;
            acc = wmma_f16(A[kc], pack16(*(const v8h*)wr, *(const v8h*)(wr + 16)), acc);
        }
        const float bv = bias[n];
        v8h outv;
        #pragma unroll
        for (int v = 0; v < 8; ++v) outv[v] = (_Float16)(acc[v] + bv);
        const size_t idx = (((size_t)mtile * 32 + nt) * 32 + lane) * 8;
        *(v8h*)(xp + idx) = outv;
    }
}

// ---------------------------------------------------------------------------
// Kernel 2: persistent LSTM recurrence. grid=(8,2), block=256.
// w_hh resident in VGPRs; h tile via LDS (f16); c in registers for all T.
// xp arrives pre-swizzled: one b128 load per gate per step.
// ---------------------------------------------------------------------------
__global__ __launch_bounds__(256)
void lstm_kernel(const _Float16* __restrict__ whhFH, const _Float16* __restrict__ whhBH,
                 const _Float16* __restrict__ xpF, const _Float16* __restrict__ xpB,
                 _Float16* __restrict__ hsF, _Float16* __restrict__ hsB)
{
    __shared__ __align__(16) _Float16 sh_h[16][HH];

    const int btile = blockIdx.x;
    const int dir   = blockIdx.y;
    const int wv    = threadIdx.x >> 5;
    const int lane  = threadIdx.x & 31;
    const int lm    = lane & 15;
    const int lg    = lane >> 4;

    const _Float16* whh = dir ? whhBH : whhFH;
    const _Float16* xp  = dir ? xpB   : xpF;
    _Float16*       hs  = dir ? hsB   : hsF;

    v16h Wb[4][4];
    #pragma unroll
    for (int g = 0; g < 4; ++g) {
        const int n = g * HH + wv * 16 + lm;
        #pragma unroll
        for (int kc = 0; kc < 4; ++kc) {
            const _Float16* wr = whh + (size_t)n * HH + kc * 32 + lg * 8;
            Wb[g][kc] = pack16(*(const v8h*)wr, *(const v8h*)(wr + 16));
        }
    }

    for (int i = threadIdx.x; i < 16 * HH; i += 256)
        ((_Float16*)sh_h)[i] = (_Float16)0.f;

    float cst[8];
    #pragma unroll
    for (int v = 0; v < 8; ++v) cst[v] = 0.f;

    const int jcol = wv * 16 + lm;
    __syncthreads();

    for (int step = 0; step < TT; ++step) {
        const int    t     = dir ? (TT - 1 - step) : step;
        const int    mt    = t * 8 + btile;
        const size_t rbase = (size_t)t * BB + btile * 16;

        if (step + 1 < TT) {    // gfx1250 global_prefetch of next step's xp
            const int tn = dir ? (TT - 2 - step) : (step + 1);
            __builtin_prefetch(xp + (((size_t)(tn * 8 + btile) * 32 + wv) * 32 + lane) * 8, 0, 0);
        }

        v16h Ah[4];
        #pragma unroll
        for (int kc = 0; kc < 4; ++kc) {
            v8h lo = *(const v8h*)&sh_h[lm][kc * 32 + lg * 8];
            v8h hi = *(const v8h*)&sh_h[lm][kc * 32 + 16 + lg * 8];
            Ah[kc] = pack16(lo, hi);
        }

        v8f acc[4];
        #pragma unroll
        for (int g = 0; g < 4; ++g) {
            const size_t idx = (((size_t)mt * 32 + (g * 8 + wv)) * 32 + lane) * 8;
            v8h xv = *(const v8h*)(xp + idx);
            #pragma unroll
            for (int v = 0; v < 8; ++v) acc[g][v] = (float)xv[v];
        }

        #pragma unroll
        for (int g = 0; g < 4; ++g)
            #pragma unroll
            for (int kc = 0; kc < 4; ++kc)
                acc[g] = wmma_f16(Ah[kc], Wb[g][kc], acc[g]);

        __syncthreads();   // all reads of previous h complete

        #pragma unroll
        for (int v = 0; v < 8; ++v) {
            const float ig = 1.f / (1.f + __expf(-acc[0][v]));
            const float fg = 1.f / (1.f + __expf(-acc[1][v]));
            const float gg = tanhf(acc[2][v]);
            const float og = 1.f / (1.f + __expf(-acc[3][v]));
            const float c  = fg * cst[v] + ig * gg;
            cst[v] = c;
            const float h = og * tanhf(c);
            const int mv = v + 8 * lg;
            sh_h[mv][jcol] = (_Float16)h;
            hs[(rbase + mv) * HH + jcol] = (_Float16)h;
        }
        __syncthreads();   // new h visible for next step
    }
}

// ---------------------------------------------------------------------------
// Kernel 3: emissions = [h_f | h_b] @ w_out^T + b_out via WMMA (TAG 9->16).
// grid=512, block=256 (8 waves), wave handles one m-tile; K=256 -> 8 WMMAs.
// ---------------------------------------------------------------------------
__global__ __launch_bounds__(256)
void emis_kernel(const _Float16* __restrict__ hsF, const _Float16* __restrict__ hsB,
                 const _Float16* __restrict__ woutP, const float* __restrict__ b_out,
                 float* __restrict__ emis)
{
    const int mtile = blockIdx.x * 8 + (threadIdx.x >> 5);
    const int lane  = threadIdx.x & 31;
    const int lm    = lane & 15;
    const int lg    = lane >> 4;
    const int r0    = mtile * 16;

    v8f acc = {};
    #pragma unroll
    for (int kc = 0; kc < 8; ++kc) {
        const int kcl = kc & 3;
        const _Float16* row = ((kc < 4) ? hsF : hsB) + (size_t)(r0 + lm) * HH + kcl * 32;
        v16h Av = pack16(*(const v8h*)(row + lg * 8), *(const v8h*)(row + 16 + lg * 8));
        const _Float16* wr = woutP + (size_t)lm * 256 + kc * 32 + lg * 8;
        v16h Bv = pack16(*(const v8h*)wr, *(const v8h*)(wr + 16));
        acc = wmma_f16(Av, Bv, acc);
    }
    if (lm < NTAG) {
        const float bv = b_out[lm];
        #pragma unroll
        for (int v = 0; v < 8; ++v)
            emis[(size_t)(r0 + v + 8 * lg) * NTAG + lm] = acc[v] + bv;
    }
}

// ---------------------------------------------------------------------------
// Kernel 4: CRF score + forward algorithm + mean NLL. 128 threads, 1 block.
// Uses exp(trans) precomputed in LDS: 9 exp + 9 log + 81 fma per step.
// ---------------------------------------------------------------------------
__global__ __launch_bounds__(128)
void crf_kernel(const int* __restrict__ tags, const unsigned char* __restrict__ mask,
                const float* __restrict__ emis,
                const float* __restrict__ start_trans, const float* __restrict__ end_trans,
                const float* __restrict__ trans, float* __restrict__ out)
{
    __shared__ float tr[NTAG][NTAG];
    __shared__ float etr[NTAG][NTAG];
    __shared__ float st[NTAG], et[NTAG];
    __shared__ float red[BB];
    const int tid = threadIdx.x;
    if (tid < NTAG * NTAG) {
        float v = trans[tid];
        ((float*)tr)[tid]  = v;
        ((float*)etr)[tid] = __expf(v);
    }
    if (tid >= 96  && tid < 96  + NTAG) st[tid - 96]  = start_trans[tid - 96];
    if (tid >= 112 && tid < 112 + NTAG) et[tid - 112] = end_trans[tid - 112];
    __syncthreads();

    const int b = tid;
    float alpha[NTAG];
    const int tg0 = tags[b * TT];
    #pragma unroll
    for (int j = 0; j < NTAG; ++j) alpha[j] = st[j] + emis[b * NTAG + j];
    float score = st[tg0] + emis[b * NTAG + tg0];
    int prev = tg0, last = tg0;

    for (int t = 1; t < TT; ++t) {
        const float* e = emis + ((size_t)t * BB + b) * NTAG;
        const int cur = tags[b * TT + t];
        const float m = mask[b * TT + t] ? 1.f : 0.f;
        score += (tr[prev][cur] + e[cur]) * m;

        float amx = alpha[0];
        #pragma unroll
        for (int i = 1; i < NTAG; ++i) amx = fmaxf(amx, alpha[i]);
        float ea[NTAG];
        #pragma unroll
        for (int i = 0; i < NTAG; ++i) ea[i] = __expf(alpha[i] - amx);

        float nxt[NTAG];
        #pragma unroll
        for (int j = 0; j < NTAG; ++j) {
            float s = 0.f;
            #pragma unroll
            for (int i = 0; i < NTAG; ++i) s += ea[i] * etr[i][j];
            nxt[j] = amx + __logf(s) + e[j];
        }
        if (m > 0.f) {
            #pragma unroll
            for (int j = 0; j < NTAG; ++j) alpha[j] = nxt[j];
            last = cur;
        }
        prev = cur;
    }

    score += et[last];
    float mx = alpha[0] + et[0];
    #pragma unroll
    for (int j = 1; j < NTAG; ++j) mx = fmaxf(mx, alpha[j] + et[j]);
    float s = 0.f;
    #pragma unroll
    for (int j = 0; j < NTAG; ++j) s += __expf(alpha[j] + et[j] - mx);
    const float logZ = mx + __logf(s);

    red[b] = logZ - score;
    __syncthreads();
    if (b == 0) {
        float tot = 0.f;
        for (int i = 0; i < BB; ++i) tot += red[i];
        out[0] = tot / (float)BB;
    }
}

// ---------------------------------------------------------------------------
extern "C" void kernel_launch(void* const* d_in, const int* in_sizes, int n_in,
                              void* d_out, int out_size, void* d_ws, size_t ws_size,
                              hipStream_t stream)
{
    const int*           sentence = (const int*)d_in[0];
    const int*           tags     = (const int*)d_in[1];
    const unsigned char* mask     = (const unsigned char*)d_in[2];
    const float*         emb      = (const float*)d_in[3];
    const float*         w_ih_f   = (const float*)d_in[4];
    const float*         w_hh_f   = (const float*)d_in[5];
    const float*         b_f      = (const float*)d_in[6];
    const float*         w_ih_b   = (const float*)d_in[7];
    const float*         w_hh_b   = (const float*)d_in[8];
    const float*         b_b      = (const float*)d_in[9];
    const float*         w_out    = (const float*)d_in[10];
    const float*         b_out    = (const float*)d_in[11];
    const float*         s_tr     = (const float*)d_in[12];
    const float*         e_tr     = (const float*)d_in[13];
    const float*         trans    = (const float*)d_in[14];

    const size_t NR = (size_t)TT * BB;   // 65536
    char* p = (char*)d_ws;
    _Float16* xpF  = (_Float16*)p; p += NR * G4 * sizeof(_Float16);
    _Float16* xpB  = (_Float16*)p; p += NR * G4 * sizeof(_Float16);
    _Float16* hsF  = (_Float16*)p; p += NR * HH * sizeof(_Float16);
    _Float16* hsB  = (_Float16*)p; p += NR * HH * sizeof(_Float16);
    float*    emis = (float*)p;    p += NR * NTAG * sizeof(float);
    _Float16* embH = (_Float16*)p; p += (size_t)VV * EE * sizeof(_Float16);
    _Float16* wihFH = (_Float16*)p; p += (size_t)G4 * EE * sizeof(_Float16);
    _Float16* wihBH = (_Float16*)p; p += (size_t)G4 * EE * sizeof(_Float16);
    _Float16* whhFH = (_Float16*)p; p += (size_t)G4 * HH * sizeof(_Float16);
    _Float16* whhBH = (_Float16*)p; p += (size_t)G4 * HH * sizeof(_Float16);
    _Float16* woutP = (_Float16*)p; p += (size_t)16 * 256 * sizeof(_Float16);

    const int nEmb = VV * EE, nW = G4 * EE;
    cvt_f16_kernel<<<(nEmb + 255) / 256, 256, 0, stream>>>(emb, embH, nEmb);
    cvt_f16_kernel<<<(nW + 255) / 256, 256, 0, stream>>>(w_ih_f, wihFH, nW);
    cvt_f16_kernel<<<(nW + 255) / 256, 256, 0, stream>>>(w_ih_b, wihBH, nW);
    cvt_f16_kernel<<<(nW + 255) / 256, 256, 0, stream>>>(w_hh_f, whhFH, nW);
    cvt_f16_kernel<<<(nW + 255) / 256, 256, 0, stream>>>(w_hh_b, whhBH, nW);
    wout_pad_kernel<<<16, 256, 0, stream>>>(w_out, woutP);

    proj_kernel<<<dim3((TT * BB) / 16, 2), 256, 0, stream>>>(
        sentence, embH, wihFH, b_f, wihBH, b_b, xpF, xpB);
    lstm_kernel<<<dim3(BB / 16, 2), 256, 0, stream>>>(
        whhFH, whhBH, xpF, xpB, hsF, hsB);
    emis_kernel<<<(TT * BB) / 16 / 8, 256, 0, stream>>>(hsF, hsB, woutP, b_out, emis);
    crf_kernel<<<1, BB, 0, stream>>>(tags, mask, emis, s_tr, e_tr, trans, (float*)d_out);
}
